// SupplyChainGNN_7069516169663
// MI455X (gfx1250) — matrix operand order repaired and checked by reading, not compiled, not run on hardware
//
#include <hip/hip_runtime.h>

// ---------------------------------------------------------------------------
// SupplyChainGNN on MI455X (gfx1250): 2-layer GCN + sigmoid head.
//   layer: h = x@W (WMMA f16->f32), agg = D^-1/2 A_hat D^-1/2 h, +b, relu
//   head : sigmoid(h2 @ Wl + bl)
// GEMM via V_WMMA_F32_16X16X32_F16; aggregation via L2-resident f32 atomics.
// ---------------------------------------------------------------------------

typedef __attribute__((ext_vector_type(16))) _Float16 v16h;
typedef __attribute__((ext_vector_type(8)))  _Float16 v8h;
typedef __attribute__((ext_vector_type(8)))  float    v8f;
typedef __attribute__((ext_vector_type(4)))  float    v4f;

#define FDIM 128   // feature width of both GCN layers

// ---- degree / normalization -------------------------------------------------
__global__ void k_deg_init(float* __restrict__ dinv, int np) {
  int i = blockIdx.x * blockDim.x + threadIdx.x;
  if (i < np) dinv[i] = 1.0f;                       // self-loop contributes 1
}

__global__ void k_deg_count(const int* __restrict__ col, float* __restrict__ dinv, int E) {
  int e = blockIdx.x * blockDim.x + threadIdx.x;
  if (e < E) unsafeAtomicAdd(&dinv[col[e]], 1.0f);  // integer-valued: exact & order-free
}

__global__ void k_rsqrt(float* __restrict__ dinv, int np) {
  int i = blockIdx.x * blockDim.x + threadIdx.x;
  if (i < np) dinv[i] = rsqrtf(dinv[i]);            // deg >= 1 always
}

// ---- f32 -> f16 conversions -------------------------------------------------
__global__ void k_cvt_x(const float* __restrict__ x, _Float16* __restrict__ xh,
                        int n_valid, int n_total) {
  int i = blockIdx.x * blockDim.x + threadIdx.x;
  if (i < n_total) xh[i] = (i < n_valid) ? (_Float16)x[i] : (_Float16)0.0f;
}

// W[k][n] (row-major KxN) -> Wt[n][k] (f16, contiguous in k)
__global__ void k_cvt_w(const float* __restrict__ W, _Float16* __restrict__ Wt) {
  int i = blockIdx.x * blockDim.x + threadIdx.x;    // i = n*128 + k
  int n = i >> 7, k = i & 127;
  Wt[i] = (_Float16)W[k * FDIM + n];
}

// ---- WMMA GEMM: hp = (xh @ W) * dinv[row]; agg initialized with self-loop ---
// block = 256 threads = 8 waves; block handles 16 rows x 128 cols,
// wave w handles the 16x16 tile at columns [16w, 16w+16).
__global__ void __launch_bounds__(256)
k_gemm_wmma(const _Float16* __restrict__ xh, const _Float16* __restrict__ Wt,
            const float* __restrict__ dinv, float* __restrict__ hp,
            float* __restrict__ agg) {
  const int lane = threadIdx.x & 31;
  const int wave = threadIdx.x >> 5;
  const int row0 = blockIdx.x * 16;
  const int n0   = wave * 16;
  const int hl   = lane & 15;
  const int kbA  = (lane < 16) ? 0 : 8;    // A 16x32 f16 layout (ISA 7.12.2)
  const int kbB  = (lane < 16) ? 0 : 8;    // B 32x16 f16 layout

  const _Float16* aptr = xh + (size_t)(row0 + hl) * FDIM;
  const _Float16* bptr = Wt + (size_t)(n0 + hl) * FDIM;

  v8f acc = {};
#pragma unroll
  for (int c = 0; c < 4; ++c) {            // K = 128 in chunks of 32
    const int kc = c * 32;
    v8h alo = *(const v8h*)(aptr + kc + kbA);
    v8h ahi = *(const v8h*)(aptr + kc + 16 + kbA);
    v16h a = __builtin_shufflevector(alo, ahi, 0,1,2,3,4,5,6,7,8,9,10,11,12,13,14,15);
    v8h blo = *(const v8h*)(bptr + kc + kbB);
    v8h bhi = *(const v8h*)(bptr + kc + 16 + kbB);
    v16h b = __builtin_shufflevector(blo, bhi, 0,1,2,3,4,5,6,7,8,9,10,11,12,13,14,15);
    acc = __builtin_amdgcn_wmma_f32_16x16x32_f16(false, a, false, b,
                                                 (short)0, acc, false, false);
  }

  // D layout: lanes 0-15 -> col n0+lane, M = r; lanes 16-31 -> M = 8 + r
  const int colD  = n0 + hl;
  const int mbase = (lane < 16) ? 0 : 8;
#pragma unroll
  for (int r = 0; r < 8; ++r) {
    const int row = row0 + mbase + r;
    const float v = acc[r] * dinv[row];    // pre-scale by dinv[row]
    hp [(size_t)row * FDIM + colD] = v;    // message source
    agg[(size_t)row * FDIM + colD] = v;    // self-loop term (dinv applied later)
  }
}

// ---- edge scatter: agg[col] += hp[row]  (one wave per edge, 4 f32/lane) -----
__global__ void __launch_bounds__(256)
k_scatter(const int* __restrict__ row, const int* __restrict__ col,
          const float* __restrict__ hp, float* __restrict__ agg, int E) {
  const int gw   = (int)((blockIdx.x * (unsigned)blockDim.x + threadIdx.x) >> 5);
  const int lane = threadIdx.x & 31;
  if (gw >= E) return;
  const int r = row[gw];
  const int c = col[gw];
  const v4f m = *(const v4f*)(hp + (size_t)r * FDIM + lane * 4);
  float* dst = agg + (size_t)c * FDIM + lane * 4;
  unsafeAtomicAdd(dst + 0, m[0]);
  unsafeAtomicAdd(dst + 1, m[1]);
  unsafeAtomicAdd(dst + 2, m[2]);
  unsafeAtomicAdd(dst + 3, m[3]);
}

// ---- post: out = relu(agg*dinv[col] + b), re-quantize to f16 for next GEMM --
__global__ void k_post(const float* __restrict__ agg, const float* __restrict__ dinv,
                       const float* __restrict__ bias, _Float16* __restrict__ xh,
                       int total) {
  int i = blockIdx.x * blockDim.x + threadIdx.x;
  if (i >= total) return;
  const int node = i >> 7, f = i & 127;
  float v = agg[i] * dinv[node] + bias[f];
  v = v > 0.0f ? v : 0.0f;
  xh[i] = (_Float16)v;
}

// ---- head: sigmoid(relu(agg*dinv + b2) . Wl + bl), one wave per node --------
__global__ void __launch_bounds__(256)
k_head(const float* __restrict__ agg, const float* __restrict__ dinv,
       const float* __restrict__ b2, const float* __restrict__ Wl,
       const float* __restrict__ bl, float* __restrict__ out, int n) {
  const int gw   = (int)((blockIdx.x * (unsigned)blockDim.x + threadIdx.x) >> 5);
  const int lane = threadIdx.x & 31;
  if (gw >= n) return;
  const float di = dinv[gw];
  const float* a = agg + (size_t)gw * FDIM;
  float s = 0.0f;
#pragma unroll
  for (int j = 0; j < 4; ++j) {
    const int f = lane + 32 * j;
    float v = a[f] * di + b2[f];
    v = v > 0.0f ? v : 0.0f;
    s += v * Wl[f];
  }
  for (int o = 16; o > 0; o >>= 1) s += __shfl_xor(s, o, 32);
  if (lane == 0) out[gw] = 1.0f / (1.0f + __expf(-(s + bl[0])));
}

// ---------------------------------------------------------------------------
extern "C" void kernel_launch(void* const* d_in, const int* in_sizes, int n_in,
                              void* d_out, int out_size, void* d_ws, size_t ws_size,
                              hipStream_t stream) {
  const float* x  = (const float*)d_in[0];
  const int*   ei = (const int*)  d_in[1];
  const float* W1 = (const float*)d_in[2];
  const float* b1 = (const float*)d_in[3];
  const float* W2 = (const float*)d_in[4];
  const float* b2 = (const float*)d_in[5];
  const float* Wl = (const float*)d_in[6];
  const float* bl = (const float*)d_in[7];
  float* out = (float*)d_out;

  const int n  = in_sizes[0] / FDIM;      // 100000
  const int E  = in_sizes[1] / 2;         // 3200000
  const int np = (n + 15) & ~15;          // rows padded to WMMA tile
  const int* erow = ei;
  const int* ecol = ei + E;

  // workspace carve-out (256B aligned chunks)
  char* ws = (char*)d_ws;
  size_t off = 0;
  auto carve = [&](size_t bytes) -> void* {
    void* p = ws + off;
    off += (bytes + 255) & ~(size_t)255;
    return p;
  };
  float*    dinv = (float*)   carve((size_t)np * 4);
  float*    hp   = (float*)   carve((size_t)np * FDIM * 4);
  float*    agg  = (float*)   carve((size_t)np * FDIM * 4);
  _Float16* xh   = (_Float16*)carve((size_t)np * FDIM * 2);
  _Float16* Wt   = (_Float16*)carve((size_t)FDIM * FDIM * 2);
  (void)ws_size; (void)n_in; (void)out_size;

  const int T = 256;
  const int totF = np * FDIM;

  // degrees -> dinv
  k_deg_init <<<(np + T - 1) / T, T, 0, stream>>>(dinv, np);
  k_deg_count<<<(E  + T - 1) / T, T, 0, stream>>>(ecol, dinv, E);
  k_rsqrt    <<<(np + T - 1) / T, T, 0, stream>>>(dinv, np);

  // layer 1
  k_cvt_x    <<<(totF + T - 1) / T, T, 0, stream>>>(x, xh, n * FDIM, totF);
  k_cvt_w    <<<(FDIM * FDIM) / T,  T, 0, stream>>>(W1, Wt);
  k_gemm_wmma<<<np / 16, 256, 0, stream>>>(xh, Wt, dinv, hp, agg);
  k_scatter  <<<(E + 7) / 8, 256, 0, stream>>>(erow, ecol, hp, agg, E);
  k_post     <<<(totF + T - 1) / T, T, 0, stream>>>(agg, dinv, b1, xh, totF);

  // layer 2
  k_cvt_w    <<<(FDIM * FDIM) / T, T, 0, stream>>>(W2, Wt);
  k_gemm_wmma<<<np / 16, 256, 0, stream>>>(xh, Wt, dinv, hp, agg);
  k_scatter  <<<(E + 7) / 8, 256, 0, stream>>>(erow, ecol, hp, agg, E);

  // linear head + sigmoid
  k_head     <<<(n + 7) / 8, 256, 0, stream>>>(agg, dinv, b2, Wl, bl, out, n);
}
